// ICLearning_91250875171329
// MI455X (gfx1250) — matrix operand order repaired and checked by reading, not compiled, not run on hardware
//
#include <hip/hip_runtime.h>
#include <math.h>

#define B_     4
#define T_     2048
#define TRAIN_ 1536
#define D_     512
#define H_     8
#define DH_    64
#define L_     12
#define FF_    2048
#define V_     100

typedef __bf16 bf16;
typedef __attribute__((ext_vector_type(16))) __bf16 v16bf;
typedef __attribute__((ext_vector_type(8)))  __bf16 v8bf;
typedef __attribute__((ext_vector_type(8)))  float  v8f;

__device__ __forceinline__ v16bf cat16(v8bf lo, v8bf hi) {
    return __builtin_shufflevector(lo, hi, 0,1,2,3,4,5,6,7,8,9,10,11,12,13,14,15);
}

// A-fragment (16x32 bf16): lane holds row M=lane%16;
// elements 0..15 = K[k0 + 16*half + 0..15] (contiguous, matches B fragments)
__device__ __forceinline__ v16bf ldA32(const bf16* rowk0, int half) {
    v8bf lo = *(const v8bf*)(rowk0 + half * 16);
    v8bf hi = *(const v8bf*)(rowk0 + half * 16 + 8);
    return cat16(lo, hi);
}

__device__ __forceinline__ v8f wmma_bf16(v16bf a, v16bf b, v8f c) {
    return __builtin_amdgcn_wmma_f32_16x16x32_bf16(false, a, false, b, (short)0, c, false, false);
}

__device__ __forceinline__ float gelu_f(float x) {
    return 0.5f * x * (1.f + tanhf(0.7978845608028654f * (x + 0.044715f * x * x * x)));
}

__device__ __forceinline__ v8f zero8() {
    v8f z = {0.f,0.f,0.f,0.f,0.f,0.f,0.f,0.f};
    return z;
}

// ---------------------------------------------------------------------------
// Pack f32 weight W[K][N] row-major into WMMA B-fragment tile layout:
// tile (kt,nt) of 32x16, 512 bf16 per tile; within tile: lane = (k%32/16)*16 + n%16,
// element e = k%16. Each lane's 16 values are contiguous -> coalesced 32B loads.
// ---------------------------------------------------------------------------
__global__ void pack_b_kernel(const float* __restrict__ W, bf16* __restrict__ Bp,
                              int K, int N, int Npad) {
    long idx = (long)blockIdx.x * 256 + threadIdx.x;
    long total = (long)K * Npad;
    if (idx >= total) return;
    int tile   = (int)(idx >> 9);
    int within = (int)(idx & 511);
    int lane = within >> 4;
    int e    = within & 15;
    int ntiles = Npad >> 4;
    int kt = tile / ntiles, nt = tile % ntiles;
    int k = kt * 32 + (lane >> 4) * 16 + e;
    int n = nt * 16 + (lane & 15);
    float v = (n < N && k < K) ? W[(size_t)k * N + n] : 0.f;
    Bp[idx] = (bf16)v;
}

// ---------------------------------------------------------------------------
// rep = R + concat(emb[y_train], 0)
// ---------------------------------------------------------------------------
__global__ void embed_add_kernel(const float* __restrict__ R, const int* __restrict__ ytr,
                                 const float* __restrict__ emb, float* __restrict__ X) {
    int idx = blockIdx.x * 256 + threadIdx.x;
    if (idx >= B_ * T_ * D_) return;
    int d  = idx & (D_ - 1);
    int bt = idx >> 9;
    int t  = bt & (T_ - 1);
    int bb = bt >> 11;
    float v = R[idx];
    if (t < TRAIN_) v += emb[(size_t)ytr[bb * TRAIN_ + t] * D_ + d];
    X[idx] = v;
}

// ---------------------------------------------------------------------------
// LayerNorm: one wave per row of 512, output bf16 (GEMM A operand).
// ---------------------------------------------------------------------------
__global__ __launch_bounds__(128) void ln_kernel(const float* __restrict__ X,
                                                 const float* __restrict__ g,
                                                 const float* __restrict__ be,
                                                 bf16* __restrict__ Y, int rows) {
    int lane = threadIdx.x & 31, wid = threadIdx.x >> 5;
    int row = blockIdx.x * 4 + wid;
    if (row >= rows) return;
    const float4* xp = (const float4*)(X + (size_t)row * D_);
    float4 r[4];
    float s = 0.f, s2 = 0.f;
#pragma unroll
    for (int j = 0; j < 4; ++j) {
        r[j] = xp[j * 32 + lane];
        s  += r[j].x + r[j].y + r[j].z + r[j].w;
        s2 += r[j].x * r[j].x + r[j].y * r[j].y + r[j].z * r[j].z + r[j].w * r[j].w;
    }
#pragma unroll
    for (int off = 16; off >= 1; off >>= 1) {
        s  += __shfl_xor(s, off, 32);
        s2 += __shfl_xor(s2, off, 32);
    }
    float mu  = s * (1.f / D_);
    float var = fmaxf(s2 * (1.f / D_) - mu * mu, 0.f);
    float rs  = rsqrtf(var + 1e-5f);
#pragma unroll
    for (int j = 0; j < 4; ++j) {
        int c0 = (j * 32 + lane) * 4;
        float vals[4] = {r[j].x, r[j].y, r[j].z, r[j].w};
#pragma unroll
        for (int c = 0; c < 4; ++c) {
            float y = (vals[c] - mu) * rs * g[c0 + c] + be[c0 + c];
            Y[(size_t)row * D_ + c0 + c] = (bf16)y;
        }
    }
}

// ---------------------------------------------------------------------------
// RoPE (interleaved) + repack: qkv f32 [B,T,3D] -> Q,K bf16 [B,H,T,DH], V^T bf16 [B,H,DH,T].
// 1/sqrt(DH)=0.125 score scale folded into Q (exact exponent shift in bf16).
// ---------------------------------------------------------------------------
__global__ void rope_kernel(const float* __restrict__ QKV,
                            bf16* __restrict__ Qo, bf16* __restrict__ Ko,
                            bf16* __restrict__ VtO) {
    int idx = blockIdx.x * 256 + threadIdx.x;
    if (idx >= B_ * T_ * H_ * (DH_ / 2)) return;
    int i  = idx & 31;
    int hh = (idx >> 5) & 7;
    int t  = (idx >> 8) & (T_ - 1);
    int bb = idx >> 19;
    size_t base = (size_t)(bb * T_ + t) * (3 * D_);
    int off = hh * DH_ + 2 * i;
    float q1 = QKV[base + off],            q2 = QKV[base + off + 1];
    float k1 = QKV[base + D_ + off],       k2 = QKV[base + D_ + off + 1];
    float v1 = QKV[base + 2 * D_ + off],   v2 = QKV[base + 2 * D_ + off + 1];
    float ex   = (float)(2 * i) * (1.f / DH_);
    float invf = expf(-11.512925464970229f * ex);   // ROPE_BASE^(-2i/DH)
    float ang  = (float)t * invf;
    float sn, cs;
    sincosf(ang, &sn, &cs);
    size_t qkb = ((size_t)(bb * H_ + hh) * T_ + t) * DH_ + 2 * i;
    Qo[qkb]     = (bf16)(0.125f * (q1 * cs - q2 * sn));
    Qo[qkb + 1] = (bf16)(0.125f * (q1 * sn + q2 * cs));
    Ko[qkb]     = (bf16)(k1 * cs - k2 * sn);
    Ko[qkb + 1] = (bf16)(k1 * sn + k2 * cs);
    size_t vtb = ((size_t)(bb * H_ + hh) * DH_ + 2 * i) * T_ + t;
    VtO[vtb]      = (bf16)v1;
    VtO[vtb + T_] = (bf16)v2;
}

// ---------------------------------------------------------------------------
// Flash attention: one wave per (b, h, 16-query tile).
// Mask: key allowed iff s < TRAIN, plus s==t for test queries (extra diag tile).
// QK^T and P*V via v_wmma_f32_16x16x32_bf16; online softmax in f32;
// P tile transposed C-layout -> A-layout through LDS (in-order DS within wave).
// ---------------------------------------------------------------------------
__global__ __launch_bounds__(128) void attn_kernel(const bf16* __restrict__ Q,
                                                   const bf16* __restrict__ Kc,
                                                   const bf16* __restrict__ Vt,
                                                   bf16* __restrict__ Ob) {
    __shared__ __align__(32) bf16 lds[4][16 * 40];
    const int lane = threadIdx.x & 31;
    const int wid  = threadIdx.x >> 5;
    const int half = lane >> 4;
    const int l16  = lane & 15;
    int task = blockIdx.x * 4 + wid;
    const int qt = task & 127;
    const int hh = (task >> 7) & 7;
    const int bb = task >> 10;
    const int qbase = qt << 4;

    const bf16* qp = Q  + ((size_t)(bb * H_ + hh) * T_) * DH_;
    const bf16* kp = Kc + ((size_t)(bb * H_ + hh) * T_) * DH_;
    const bf16* vp = Vt + ((size_t)(bb * H_ + hh) * DH_) * T_;
    bf16* myl = &lds[wid][0];

    const int qrow = qbase + l16;
    v16bf qa0 = ldA32(qp + (size_t)qrow * DH_,      half);  // dh 0..31
    v16bf qa1 = ldA32(qp + (size_t)qrow * DH_ + 32, half);  // dh 32..63

    v8f O[4];
#pragma unroll
    for (int j = 0; j < 4; ++j) O[j] = zero8();
    float mrow[8], lrow[8];
#pragma unroll
    for (int v = 0; v < 8; ++v) { mrow[v] = -INFINITY; lrow[v] = 0.f; }

    const bool hasDiag = (qbase >= TRAIN_);
    const int nIter = (TRAIN_ / 32) + (hasDiag ? 1 : 0);
    for (int it = 0; it < nIter; ++it) {
        const bool diag = (it == TRAIN_ / 32);
        const int skey = diag ? qbase : it * 32;

        // S = Q * K^T for 32 keys (two 16-key subtiles, K-dim = DH = 2x32)
        v8f S0 = zero8(), S1 = zero8();
        {
            int r0 = skey + l16;      if (r0 > T_ - 1) r0 = T_ - 1;
            int r1 = skey + 16 + l16; if (r1 > T_ - 1) r1 = T_ - 1;
            v16bf kb;
            kb = *(const v16bf*)(kp + (size_t)r0 * DH_ + half * 16);      S0 = wmma_bf16(qa0, kb, S0);
            kb = *(const v16bf*)(kp + (size_t)r0 * DH_ + 32 + half * 16); S0 = wmma_bf16(qa1, kb, S0);
            kb = *(const v16bf*)(kp + (size_t)r1 * DH_ + half * 16);      S1 = wmma_bf16(qa0, kb, S1);
            kb = *(const v16bf*)(kp + (size_t)r1 * DH_ + 32 + half * 16); S1 = wmma_bf16(qa1, kb, S1);
        }
        if (diag) {
#pragma unroll
            for (int v = 0; v < 8; ++v) {
                S0[v] = (l16 == v + 8 * half) ? S0[v] : -INFINITY;
                S1[v] = -INFINITY;
            }
        }

        // online softmax; C-layout row m = v + 8*half, replicated across 16 lanes/half
        float mnew[8], alpha[8], rsum[8];
#pragma unroll
        for (int v = 0; v < 8; ++v) {
            float mx = fmaxf(S0[v], S1[v]);
#pragma unroll
            for (int off = 1; off < 16; off <<= 1) mx = fmaxf(mx, __shfl_xor(mx, off, 32));
            mnew[v]  = fmaxf(mrow[v], mx);
            alpha[v] = __expf(mrow[v] - mnew[v]);
        }
#pragma unroll
        for (int v = 0; v < 8; ++v) {
            float p0 = __expf(S0[v] - mnew[v]);
            float p1 = __expf(S1[v] - mnew[v]);
            S0[v] = p0; S1[v] = p1;
            float rs = p0 + p1;
#pragma unroll
            for (int off = 1; off < 16; off <<= 1) rs += __shfl_xor(rs, off, 32);
            rsum[v] = rs;
        }
#pragma unroll
        for (int v = 0; v < 8; ++v) {
            lrow[v] = lrow[v] * alpha[v] + rsum[v];
            mrow[v] = mnew[v];
        }
#pragma unroll
        for (int j = 0; j < 4; ++j)
#pragma unroll
            for (int v = 0; v < 8; ++v) O[j][v] *= alpha[v];

        // P (16x32) C-layout -> LDS row-major (stride 40) -> A fragment
#pragma unroll
        for (int v = 0; v < 8; ++v) {
            int m = v + 8 * half;
            myl[m * 40 + l16]      = (bf16)S0[v];
            myl[m * 40 + 16 + l16] = (bf16)S1[v];
        }
        v8bf plo = *(const v8bf*)(myl + l16 * 40 + half * 16);
        v8bf phi = *(const v8bf*)(myl + l16 * 40 + half * 16 + 8);
        v16bf pa = cat16(plo, phi);

        // O += P * V  (B frag from V^T: contiguous 16 keys per lane)
        int s0 = skey + half * 16; if (s0 > T_ - 16) s0 = T_ - 16;
#pragma unroll
        for (int j = 0; j < 4; ++j) {
            v16bf vb = *(const v16bf*)(vp + (size_t)(j * 16 + l16) * T_ + s0);
            O[j] = wmma_bf16(pa, vb, O[j]);
        }
    }

    // normalize and store to o (bf16, [B,T,D] row-major -> A of Wo GEMM)
#pragma unroll
    for (int j = 0; j < 4; ++j)
#pragma unroll
        for (int v = 0; v < 8; ++v) {
            int m = v + 8 * half;
            float o = O[j][v] / lrow[v];
            Ob[(size_t)(bb * T_ + qbase + m) * D_ + hh * DH_ + j * 16 + l16] = (bf16)o;
        }
}

// ---------------------------------------------------------------------------
// GEMM: C[M,Nreal] = A[M,K](bf16,row-major) * Bp(packed bf16) + bias, epilogues:
//   mode 0: f32 out = acc + bias
//   mode 1: f32 out = res + acc + bias
//   mode 2: bf16 out = gelu(acc + bias)
// Block = 4 waves; each wave computes a 32x64 strip (2 A frags x 4 B frags,
// 8 accumulators) so every B fragment load feeds two WMMAs. Block tile 128x64.
// M must be a multiple of 128 (8192 and 2048 here).
// ---------------------------------------------------------------------------
__global__ __launch_bounds__(128) void gemm_bf16_kernel(
    const bf16* __restrict__ A, const bf16* __restrict__ Bp,
    const float* __restrict__ bias, const float* __restrict__ res,
    float* __restrict__ Of, bf16* __restrict__ Obf,
    int M, int K, int Npad, int Nreal, int mode) {
    const int lane = threadIdx.x & 31;
    const int wid  = threadIdx.x >> 5;
    const int half = lane >> 4;
    const int l16  = lane & 15;
    const int mbase  = blockIdx.y * 128 + wid * 32;
    const int nblock = blockIdx.x * 64;
    const int ntiles = Npad >> 4;

    v8f acc0[4], acc1[4];
#pragma unroll
    for (int j = 0; j < 4; ++j) { acc0[j] = zero8(); acc1[j] = zero8(); }
    bool nv[4];
#pragma unroll
    for (int j = 0; j < 4; ++j) nv[j] = (nblock + j * 16) < Npad;

    const bf16* arow0 = A + (size_t)(mbase + l16) * K;
    const bf16* arow1 = A + (size_t)(mbase + 16 + l16) * K;
    const int ktn = K >> 5;
    for (int kt = 0; kt < ktn; ++kt) {
        v16bf af0 = ldA32(arow0 + kt * 32, half);
        v16bf af1 = ldA32(arow1 + kt * 32, half);
        if (kt + 1 < ktn) {
            __builtin_prefetch(arow0 + (kt + 1) * 32, 0, 1);
            __builtin_prefetch(arow1 + (kt + 1) * 32, 0, 1);
        }
#pragma unroll
        for (int j = 0; j < 4; ++j) {
            if (!nv[j]) continue;
            int nt = (nblock >> 4) + j;
            v16bf bfv = *(const v16bf*)(Bp + ((size_t)(kt * ntiles + nt) << 9) + lane * 16);
            acc0[j] = wmma_bf16(af0, bfv, acc0[j]);
            acc1[j] = wmma_bf16(af1, bfv, acc1[j]);
        }
    }
#pragma unroll
    for (int j = 0; j < 4; ++j) {
        if (!nv[j]) continue;
        int n = nblock + j * 16 + l16;
        if (n >= Nreal) continue;
        float bb = bias[n];
#pragma unroll
        for (int g = 0; g < 2; ++g) {
#pragma unroll
            for (int v = 0; v < 8; ++v) {
                int m = mbase + g * 16 + v + 8 * half;
                size_t o = (size_t)m * Nreal + n;
                float val = (g == 0 ? acc0[j][v] : acc1[j][v]) + bb;
                if (mode == 1) val += res[o];
                if (mode == 2) Obf[o] = (bf16)gelu_f(val);
                else           Of[o]  = val;
            }
        }
    }
}

// ---------------------------------------------------------------------------
// Gather test tokens (t >= TRAIN) into bf16 matrix [B*512, D] for head GEMM.
// ---------------------------------------------------------------------------
__global__ void gather_test_kernel(const float* __restrict__ X, bf16* __restrict__ Y) {
    int idx = blockIdx.x * 256 + threadIdx.x;
    if (idx >= B_ * (T_ - TRAIN_) * D_) return;
    int d  = idx & (D_ - 1);
    int r  = idx >> 9;
    int tt = r & 511;
    int bb = r >> 9;
    Y[idx] = (bf16)X[(size_t)(bb * T_ + TRAIN_ + tt) * D_ + d];
}

// ---------------------------------------------------------------------------
extern "C" void kernel_launch(void* const* d_in, const int* in_sizes, int n_in,
                              void* d_out, int out_size, void* d_ws, size_t ws_size,
                              hipStream_t stream) {
    (void)in_sizes; (void)n_in; (void)out_size; (void)ws_size;
    const float* R    = (const float*)d_in[0];
    const int*   ytr  = (const int*)  d_in[1];
    const float* emb  = (const float*)d_in[2];
    const float* Wqkv = (const float*)d_in[3];
    const float* bqkv = (const float*)d_in[4];
    const float* Wo   = (const float*)d_in[5];
    const float* bo   = (const float*)d_in[6];
    const float* ln1g = (const float*)d_in[7];
    const float* ln1b = (const float*)d_in[8];
    const float* ln2g = (const float*)d_in[9];
    const float* ln2b = (const float*)d_in[10];
    const float* W1   = (const float*)d_in[11];
    const float* b1   = (const float*)d_in[12];
    const float* W2   = (const float*)d_in[13];
    const float* b2   = (const float*)d_in[14];
    const float* pW1  = (const float*)d_in[15];
    const float* pb1  = (const float*)d_in[16];
    const float* pW2  = (const float*)d_in[17];
    const float* pb2  = (const float*)d_in[18];
    float* out = (float*)d_out;

    char* cur = (char*)d_ws;
    auto alloc = [&](size_t bytes) -> char* {
        char* p = cur;
        cur += (bytes + 255) & ~(size_t)255;
        return p;
    };
    const size_t MT = (size_t)B_ * T_;           // 8192 rows
    float* x   = (float*)alloc(MT * D_ * 4);
    bf16*  h   = (bf16*) alloc(MT * D_ * 2);
    float* qkv = (float*)alloc(MT * 3 * D_ * 4); // reused: u (bf16), t1 (bf16)
    bf16*  u   = (bf16*)qkv;
    bf16*  t1  = (bf16*)qkv;
    bf16*  qb  = (bf16*) alloc(MT * D_ * 2);
    bf16*  kb  = (bf16*) alloc(MT * D_ * 2);
    bf16*  vT  = (bf16*) alloc(MT * D_ * 2);
    bf16*  ob  = (bf16*) alloc(MT * D_ * 2);
    bf16* pQKVb = (bf16*)alloc((size_t)L_ * D_ * 3 * D_ * 2);
    bf16* pWob  = (bf16*)alloc((size_t)L_ * D_ * D_ * 2);
    bf16* pW1b  = (bf16*)alloc((size_t)L_ * D_ * FF_ * 2);
    bf16* pW2b  = (bf16*)alloc((size_t)L_ * FF_ * D_ * 2);
    bf16* pHW1  = (bf16*)alloc((size_t)D_ * (2 * D_) * 2);
    bf16* pHW2  = (bf16*)alloc((size_t)(2 * D_) * 112 * 2);

    auto pgrid = [](size_t elems) { return dim3((unsigned)((elems + 255) / 256)); };
    for (int l = 0; l < L_; ++l) {
        pack_b_kernel<<<pgrid((size_t)D_ * 3 * D_), 256, 0, stream>>>(
            Wqkv + (size_t)l * D_ * 3 * D_, pQKVb + (size_t)l * D_ * 3 * D_, D_, 3 * D_, 3 * D_);
        pack_b_kernel<<<pgrid((size_t)D_ * D_), 256, 0, stream>>>(
            Wo + (size_t)l * D_ * D_, pWob + (size_t)l * D_ * D_, D_, D_, D_);
        pack_b_kernel<<<pgrid((size_t)D_ * FF_), 256, 0, stream>>>(
            W1 + (size_t)l * D_ * FF_, pW1b + (size_t)l * D_ * FF_, D_, FF_, FF_);
        pack_b_kernel<<<pgrid((size_t)FF_ * D_), 256, 0, stream>>>(
            W2 + (size_t)l * FF_ * D_, pW2b + (size_t)l * FF_ * D_, FF_, D_, D_);
    }
    pack_b_kernel<<<pgrid((size_t)D_ * 2 * D_), 256, 0, stream>>>(pW1, pHW1, D_, 2 * D_, 2 * D_);
    pack_b_kernel<<<pgrid((size_t)2 * D_ * 112), 256, 0, stream>>>(pW2, pHW2, 2 * D_, V_, 112);

    embed_add_kernel<<<(B_ * T_ * D_ + 255) / 256, 256, 0, stream>>>(R, ytr, emb, x);

    dim3 blk(128);
    for (int l = 0; l < L_; ++l) {
        ln_kernel<<<(unsigned)(MT / 4), blk, 0, stream>>>(x, ln1g + l * D_, ln1b + l * D_, h, (int)MT);
        gemm_bf16_kernel<<<dim3(3 * D_ / 64, (unsigned)(MT / 128)), blk, 0, stream>>>(
            h, pQKVb + (size_t)l * D_ * 3 * D_, bqkv + (size_t)l * 3 * D_, nullptr,
            qkv, nullptr, (int)MT, D_, 3 * D_, 3 * D_, 0);
        rope_kernel<<<(B_ * T_ * H_ * 32 + 255) / 256, 256, 0, stream>>>(qkv, qb, kb, vT);
        attn_kernel<<<(B_ * H_ * (T_ / 16)) / 4, blk, 0, stream>>>(qb, kb, vT, ob);
        gemm_bf16_kernel<<<dim3(D_ / 64, (unsigned)(MT / 128)), blk, 0, stream>>>(
            ob, pWob + (size_t)l * D_ * D_, bo + (size_t)l * D_, x,
            x, nullptr, (int)MT, D_, D_, D_, 1);
        ln_kernel<<<(unsigned)(MT / 4), blk, 0, stream>>>(x, ln2g + l * D_, ln2b + l * D_, h, (int)MT);
        gemm_bf16_kernel<<<dim3(FF_ / 64, (unsigned)(MT / 128)), blk, 0, stream>>>(
            h, pW1b + (size_t)l * D_ * FF_, b1 + (size_t)l * FF_, nullptr,
            nullptr, u, (int)MT, D_, FF_, FF_, 2);
        gemm_bf16_kernel<<<dim3(D_ / 64, (unsigned)(MT / 128)), blk, 0, stream>>>(
            u, pW2b + (size_t)l * FF_ * D_, b2 + (size_t)l * D_, x,
            x, nullptr, (int)MT, FF_, D_, D_, 1);
    }

    const int MH = B_ * (T_ - TRAIN_);  // 2048 test rows
    gather_test_kernel<<<(MH * D_ + 255) / 256, 256, 0, stream>>>(x, h);
    gemm_bf16_kernel<<<dim3(2 * D_ / 64, MH / 128), blk, 0, stream>>>(
        h, pHW1, pb1, nullptr, nullptr, t1, MH, D_, 2 * D_, 2 * D_, 2);
    gemm_bf16_kernel<<<dim3(2, MH / 128), blk, 0, stream>>>(
        t1, pHW2, pb2, nullptr, out, nullptr, MH, 2 * D_, 112, V_, 0);
}
